// MultiHeadAttention_48000554500736
// MI455X (gfx1250) — matrix operand order repaired and checked by reading, not compiled, not run on hardware
//
#include <hip/hip_runtime.h>

// ---------------------------------------------------------------------------
// MI455X (gfx1250) multi-head causal attention, bf16 WMMA + f32 accumulation.
// B=4, S=2048, D=1024, H=16, d=64.  Output = concat(out, k, v) in fp32.
// ---------------------------------------------------------------------------

#define Bv 4
#define Sv 2048
#define Dv 1024
#define Hv 16
#define dv 64
#define SCALE 0.125f   // d^-0.5 = 64^-0.5
#define KVLD 68        // padded LDS row stride (halves): 136B -> 34 banks, gcd(34,64)=2
#define PSLD 68

typedef __attribute__((ext_vector_type(16))) __bf16 v16bf;
typedef __attribute__((ext_vector_type(8)))  float  v8f;

union AFrag { unsigned u[8]; v16bf v; };

static __device__ __forceinline__ unsigned short f2bf(float x) {
  union { float f; unsigned u; } v; v.f = x;
  unsigned r = v.u + 0x7fffu + ((v.u >> 16) & 1u);   // round-to-nearest-even
  return (unsigned short)(r >> 16);
}

static __device__ __forceinline__ v8f wmma_bf16(v16bf a, v16bf b, v8f c) {
  return __builtin_amdgcn_wmma_f32_16x16x32_bf16(false, a, false, b,
                                                 (short)0, c, false, false);
}

// A fragment (16x32 bf16), source row-major [M x K], rows row..row+15, k0..k0+31.
// Contiguous k order used here (matches load_bT/load_b_rm): lanes 0-15 M=lane, VGPR i -> K=2i,2i+1 ;
// lanes 16-31 add +16 to K.
static __device__ __forceinline__ v16bf load_a(const unsigned short* base,
                                               int row, int ld, int k0, int lane) {
  AFrag f;
  int m = lane & 15, lh = lane >> 4;
  const unsigned short* p = base + (long)(row + m) * ld + k0 + lh * 16;
#pragma unroll
  for (int i = 0; i < 8; ++i) {
    int k = 2 * i;
    f.u[i] = *(const unsigned*)(p + k);
  }
  return f.v;
}

// B fragment for B = W^T, W row-major [N x K] (B[k][n] = W[n][k]).
// Same contiguous k order as load_a: lanes hold N, lane-half splits K 0-15 / 16-31, pairs packed.
static __device__ __forceinline__ v16bf load_bT(const unsigned short* W,
                                                int n0, int ld, int k0, int lane) {
  AFrag f;
  int n = lane & 15, lh = lane >> 4;
  const unsigned short* p = W + (long)(n0 + n) * ld + k0 + lh * 16;
#pragma unroll
  for (int j = 0; j < 8; ++j) f.u[j] = *(const unsigned*)(p + 2 * j);
  return f.v;
}

// B fragment for row-major B itself: B[k][n] = M[k0+k][n], K pairs strided by ld.
static __device__ __forceinline__ v16bf load_b_rm(const unsigned short* Mx,
                                                  int n0, int ld, int k0, int lane) {
  AFrag f;
  int n = n0 + (lane & 15), lh = lane >> 4;
#pragma unroll
  for (int j = 0; j < 8; ++j) {
    int k = k0 + lh * 16 + 2 * j;
    unsigned lo = Mx[k * ld + n];
    unsigned hi = Mx[(k + 1) * ld + n];
    f.u[j] = lo | (hi << 16);
  }
  return f.v;
}

// ---------------------------------------------------------------------------
// Stage 0: fp32 -> bf16 conversion
// ---------------------------------------------------------------------------
__global__ void cvt_bf16(const float* __restrict__ src,
                         unsigned short* __restrict__ dst, int n) {
  int i = blockIdx.x * blockDim.x + threadIdx.x;
  if (i < n) dst[i] = f2bf(src[i]);
}

// ---------------------------------------------------------------------------
// Stage 1: QKV GEMM  (M=8192, N=3072, K=1024).
// 256 threads = 8 waves; wave tile 64x64 (4x4 WMMA accums, 32 FLOP/B from L2);
// wave grid 2x4 -> block tile 128x256.  Scatters K/V fp32 to d_out and bf16
// Q/K/V in (B,H,S,d) layout to workspace.
// ---------------------------------------------------------------------------
__global__ __launch_bounds__(256) void qkv_gemm(
    const unsigned short* __restrict__ xb,   // [8192 x 1024]
    const unsigned short* __restrict__ wb,   // [3072 x 1024]
    unsigned short* __restrict__ qb,
    unsigned short* __restrict__ kbf,
    unsigned short* __restrict__ vbf,
    float* __restrict__ out_k,
    float* __restrict__ out_v) {
  const int K = Dv;
  int lane = threadIdx.x & 31, wave = threadIdx.x >> 5;
  int wm = wave & 1, wn = wave >> 1;
  int rowBase = blockIdx.x * 128 + wm * 64;
  int colBase = blockIdx.y * 256 + wn * 64;

  v8f acc[4][4] = {};
  for (int k0 = 0; k0 < K; k0 += 32) {
    __builtin_prefetch(xb + (long)rowBase * K + k0 + 64, 0, 0);
    __builtin_prefetch(wb + (long)colBase * K + k0 + 64, 0, 0);
    v16bf a[4], b[4];
#pragma unroll
    for (int i = 0; i < 4; ++i) a[i] = load_a(xb, rowBase + 16 * i, K, k0, lane);
#pragma unroll
    for (int j = 0; j < 4; ++j) b[j] = load_bT(wb, colBase + 16 * j, K, k0, lane);
#pragma unroll
    for (int i = 0; i < 4; ++i)
#pragma unroll
      for (int j = 0; j < 4; ++j)
        acc[i][j] = wmma_bf16(a[i], b[j], acc[i][j]);
  }

  int lh = lane >> 4;
#pragma unroll
  for (int ti = 0; ti < 4; ++ti)
#pragma unroll
    for (int tj = 0; tj < 4; ++tj)
#pragma unroll
      for (int i = 0; i < 8; ++i) {
        int row = rowBase + ti * 16 + i + lh * 8;    // global token index b*S+s
        int col = colBase + tj * 16 + (lane & 15);   // qkv feature index
        float val = acc[ti][tj][i];
        unsigned short bv = f2bf(val);
        int b = row >> 11, s = row & (Sv - 1);
        if (col < Dv) {
          int h = col >> 6, dd = col & 63;
          qb[((long)((b << 4) + h) * Sv + s) * dv + dd] = bv;
        } else if (col < 2 * Dv) {
          int e = col - Dv; int h = e >> 6, dd = e & 63;
          long idx = ((long)((b << 4) + h) * Sv + s) * dv + dd;
          kbf[idx] = bv;
          out_k[idx] = val;
        } else {
          int e = col - 2 * Dv; int h = e >> 6, dd = e & 63;
          long idx = ((long)((b << 4) + h) * Sv + s) * dv + dd;
          vbf[idx] = bv;
          out_v[idx] = val;
        }
      }
}

// ---------------------------------------------------------------------------
// Stage 2: causal flash attention.  Block = 4 waves = 64 query rows of one
// (b,h).  64-key K/V tiles staged in padded LDS (bank-conflict-free B-frag
// reads), online softmax, P re-layout through wave-private LDS, 16 WMMAs per
// barrier round.
// ---------------------------------------------------------------------------
__global__ __launch_bounds__(128) void attn_fwd(
    const unsigned short* __restrict__ qb,
    const unsigned short* __restrict__ kbf,
    const unsigned short* __restrict__ vbf,
    unsigned short* __restrict__ ab) {   // (B,S,D) bf16
  __shared__ __align__(16) unsigned short Ks[64 * KVLD];
  __shared__ __align__(16) unsigned short Vs[64 * KVLD];
  __shared__ __align__(16) unsigned short Ps[4][16 * PSLD];

  int lane = threadIdx.x & 31, wave = threadIdx.x >> 5;
  int lh = lane >> 4;
  int b = blockIdx.z, h = blockIdx.y, qt = blockIdx.x;
  long bh = (long)(b * Hv + h) * Sv * dv;
  const unsigned short* Qg = qb + bh;
  const unsigned short* Kg = kbf + bh;
  const unsigned short* Vg = vbf + bh;

  int qr0 = qt * 64 + wave * 16;
  v16bf aq0 = load_a(Qg, qr0, dv, 0, lane);
  v16bf aq1 = load_a(Qg, qr0, dv, 32, lane);

  v8f o[4] = {};
  float m_i[8], l_i[8];
#pragma unroll
  for (int i = 0; i < 8; ++i) { m_i[i] = -__builtin_inff(); l_i[i] = 0.f; }

  int nkt = qt + 1;                      // 64-key tiles, causal upper bound
  for (int kt = 0; kt < nkt; ++kt) {
    __syncthreads();
    {
      // stage 64x64 K and V tiles, 8B chunks, padded LDS rows
      const unsigned long long* sk =
          (const unsigned long long*)(Kg + (long)kt * 64 * dv);
      const unsigned long long* sv =
          (const unsigned long long*)(Vg + (long)kt * 64 * dv);
#pragma unroll
      for (int ii = 0; ii < 8; ++ii) {
        int cc = threadIdx.x + 128 * ii;      // 1024 chunks of 8B
        int r = cc >> 4, c = cc & 15;         // row 0..63, 8B-chunk 0..15
        *(unsigned long long*)(Ks + r * KVLD + c * 4) = sk[r * 16 + c];
        *(unsigned long long*)(Vs + r * KVLD + c * 4) = sv[r * 16 + c];
      }
    }
    __syncthreads();

    // scores: four 16x16 tiles covering 64 keys, S = Q(16x64) * K^T
    v8f s[4];
#pragma unroll
    for (int g = 0; g < 4; ++g) {
      v8f z = {};
      z = wmma_bf16(aq0, load_bT(Ks, g * 16, KVLD, 0, lane), z);
      z = wmma_bf16(aq1, load_bT(Ks, g * 16, KVLD, 32, lane), z);
      s[g] = z;
    }

    // scale + causal mask + online softmax (rows live in 16-lane groups)
#pragma unroll
    for (int i = 0; i < 8; ++i) {
      int q = qr0 + i + lh * 8;
      int c0 = kt * 64 + (lane & 15);
      float p[4];
      float t = -__builtin_inff();
#pragma unroll
      for (int g = 0; g < 4; ++g) {
        float v = s[g][i] * SCALE;
        if (c0 + g * 16 > q) v = -__builtin_inff();
        p[g] = v;
        t = fmaxf(t, v);
      }
      t = fmaxf(t, __shfl_xor(t, 1, 16));
      t = fmaxf(t, __shfl_xor(t, 2, 16));
      t = fmaxf(t, __shfl_xor(t, 4, 16));
      t = fmaxf(t, __shfl_xor(t, 8, 16));

      float mn = fmaxf(m_i[i], t);
      float alpha = __expf(m_i[i] - mn);
      float rs = 0.f;
#pragma unroll
      for (int g = 0; g < 4; ++g) { p[g] = __expf(p[g] - mn); rs += p[g]; }
      rs += __shfl_xor(rs, 1, 16);
      rs += __shfl_xor(rs, 2, 16);
      rs += __shfl_xor(rs, 4, 16);
      rs += __shfl_xor(rs, 8, 16);

      l_i[i] = l_i[i] * alpha + rs;
      m_i[i] = mn;
#pragma unroll
      for (int c = 0; c < 4; ++c) o[c][i] *= alpha;

      // stage P (C-layout -> row-major bf16) in wave-private LDS
      int r = i + lh * 8;
      unsigned short* pp = &Ps[wave][r * PSLD + (lane & 15)];
#pragma unroll
      for (int g = 0; g < 4; ++g) pp[g * 16] = f2bf(p[g]);
    }
    // wave-private LDS RAW fence (cross-lane through memory)
    asm volatile("s_wait_dscnt 0x0" ::: "memory");

    v16bf ap0 = load_a(&Ps[wave][0], 0, PSLD, 0, lane);
    v16bf ap1 = load_a(&Ps[wave][0], 0, PSLD, 32, lane);
#pragma unroll
    for (int c = 0; c < 4; ++c) {
      o[c] = wmma_bf16(ap0, load_b_rm(Vs, c * 16, KVLD, 0, lane), o[c]);
      o[c] = wmma_bf16(ap1, load_b_rm(Vs, c * 16, KVLD, 32, lane), o[c]);
    }
  }

  // epilogue: normalize, store bf16 attention output (B,S,D)
#pragma unroll
  for (int i = 0; i < 8; ++i) {
    float rl = 1.f / l_i[i];
    int srow = qr0 + i + lh * 8;
    long orow = ((long)b * Sv + srow) * Dv;
#pragma unroll
    for (int c = 0; c < 4; ++c) {
      int col = h * dv + c * 16 + (lane & 15);
      ab[orow + col] = f2bf(o[c][i] * rl);
    }
  }
}

// ---------------------------------------------------------------------------
// Stage 3: output projection (M=8192, N=1024, K=1024), fp32 result to d_out.
// Same 64x64 wave tiling as stage 1.
// ---------------------------------------------------------------------------
__global__ __launch_bounds__(256) void out_gemm(
    const unsigned short* __restrict__ ab,   // [8192 x 1024] bf16
    const unsigned short* __restrict__ wb,   // [1024 x 1024] bf16
    float* __restrict__ out) {
  const int K = Dv;
  int lane = threadIdx.x & 31, wave = threadIdx.x >> 5;
  int wm = wave & 1, wn = wave >> 1;
  int rowBase = blockIdx.x * 128 + wm * 64;
  int colBase = blockIdx.y * 256 + wn * 64;

  v8f acc[4][4] = {};
  for (int k0 = 0; k0 < K; k0 += 32) {
    __builtin_prefetch(ab + (long)rowBase * K + k0 + 64, 0, 0);
    __builtin_prefetch(wb + (long)colBase * K + k0 + 64, 0, 0);
    v16bf a[4], b[4];
#pragma unroll
    for (int i = 0; i < 4; ++i) a[i] = load_a(ab, rowBase + 16 * i, K, k0, lane);
#pragma unroll
    for (int j = 0; j < 4; ++j) b[j] = load_bT(wb, colBase + 16 * j, K, k0, lane);
#pragma unroll
    for (int i = 0; i < 4; ++i)
#pragma unroll
      for (int j = 0; j < 4; ++j)
        acc[i][j] = wmma_bf16(a[i], b[j], acc[i][j]);
  }

  int lh = lane >> 4;
#pragma unroll
  for (int ti = 0; ti < 4; ++ti)
#pragma unroll
    for (int tj = 0; tj < 4; ++tj)
#pragma unroll
      for (int i = 0; i < 8; ++i) {
        int row = rowBase + ti * 16 + i + lh * 8;
        int col = colBase + tj * 16 + (lane & 15);
        out[(long)row * Dv + col] = acc[ti][tj][i];
      }
}

// ---------------------------------------------------------------------------
// Host-side launcher
// ---------------------------------------------------------------------------
extern "C" void kernel_launch(void* const* d_in, const int* in_sizes, int n_in,
                              void* d_out, int out_size, void* d_ws, size_t ws_size,
                              hipStream_t stream) {
  const float* x     = (const float*)d_in[0];
  // d_in[1] = mask (bool) — causality is applied analytically, not read.
  const float* w_qkv = (const float*)d_in[2];
  const float* w_out = (const float*)d_in[3];
  float* out = (float*)d_out;

  const long nX   = (long)Bv * Sv * Dv;        // 8388608
  const long nWq  = (long)3 * Dv * Dv;         // 3145728
  const long nWo  = (long)Dv * Dv;             // 1048576

  // workspace layout (bf16 buffers), all 16B aligned
  char* ws = (char*)d_ws;
  unsigned short* xb    = (unsigned short*)(ws);
  unsigned short* wqkvb = (unsigned short*)(ws + 16777216);
  unsigned short* woutb = (unsigned short*)(ws + 23068672);
  unsigned short* qb    = (unsigned short*)(ws + 25165824);
  unsigned short* kbf   = (unsigned short*)(ws + 41943040);
  unsigned short* vbf   = (unsigned short*)(ws + 58720256);
  unsigned short* ab    = (unsigned short*)(ws + 75497472);

  float* out_k = out + nX;         // second tuple element: k (B,H,S,d)
  float* out_v = out + 2 * nX;     // third: v (B,H,S,d)

  // Stage 0: fp32 -> bf16
  cvt_bf16<<<dim3((nX  + 255) / 256), 256, 0, stream>>>(x,     xb,    (int)nX);
  cvt_bf16<<<dim3((nWq + 255) / 256), 256, 0, stream>>>(w_qkv, wqkvb, (int)nWq);
  cvt_bf16<<<dim3((nWo + 255) / 256), 256, 0, stream>>>(w_out, woutb, (int)nWo);

  // Stage 1: QKV projection + K/V scatter  (block tile 128x256)
  qkv_gemm<<<dim3(64, 12), 256, 0, stream>>>(xb, wqkvb, qb, kbf, vbf,
                                             out_k, out_v);

  // Stage 2: causal flash attention (64 q-rows x 64-key tiles per block)
  attn_fwd<<<dim3(Sv / 64, Hv, Bv), 128, 0, stream>>>(qb, kbf, vbf, ab);

  // Stage 3: output projection  (block tile 128x256)
  out_gemm<<<dim3(64, 4), 256, 0, stream>>>(ab, woutb, out);
}
